// SpatialBlock_13572096655437
// MI455X (gfx1250) — matrix operand order repaired and checked by reading, not compiled, not run on hardware
//
#include <hip/hip_runtime.h>
#include <hip/hip_bf16.h>
#include <stdint.h>

// ---------------------------------------------------------------------------
// Types for CDNA5 WMMA (gfx1250, wave32)
// ---------------------------------------------------------------------------
typedef __bf16 v16bf __attribute__((ext_vector_type(16)));
typedef float  v8f   __attribute__((ext_vector_type(8)));
typedef int    v4i   __attribute__((ext_vector_type(4)));

#define B_    4
#define D_    8
#define H_    56
#define W_    56
#define C_    256
#define HID_  1024
#define T_    (B_ * D_ * H_ * W_)   // 100352 tokens, divisible by 32
#define HEADS_ 8
#define HD_   32

__device__ __forceinline__ float b2f(unsigned short u) {
  return __uint_as_float(((unsigned int)u) << 16);
}
__device__ __forceinline__ unsigned short f2b(float f) {
  unsigned int u = __float_as_uint(f);
  u += 0x7FFFu + ((u >> 16) & 1u);          // round-to-nearest-even
  return (unsigned short)(u >> 16);
}

#if __has_builtin(__builtin_amdgcn_global_load_async_to_lds_b128)
#define HAVE_ASYNC_LDS 1
typedef __attribute__((address_space(1))) v4i* gv4i_p;   // global int4*
typedef __attribute__((address_space(3))) v4i* lv4i_p;   // LDS int4*
#else
#define HAVE_ASYNC_LDS 0
#endif

__device__ __forceinline__ void wait_async_lds() {
#if HAVE_ASYNC_LDS
#if __has_builtin(__builtin_amdgcn_s_wait_asynccnt)
  __builtin_amdgcn_s_wait_asynccnt(0);
#else
  asm volatile("s_wait_asynccnt 0x0" ::: "memory");
#endif
#endif
}

// ---------------------------------------------------------------------------
// f32 -> bf16 conversion (weights)
// ---------------------------------------------------------------------------
__global__ void cvt_f32_bf16(const float* __restrict__ in,
                             unsigned short* __restrict__ out, int n) {
  int i = blockIdx.x * blockDim.x + threadIdx.x;
  if (i < n) out[i] = f2b(in[i]);
}

// ---------------------------------------------------------------------------
// ConvPosEnc: x + dw3x3_hw(x) + dw3x3_dh(x) + dw3x3_dw(x)  (+ per-conv bias)
// One block per token (256 threads == 256 channels). SAME zero padding.
// w layout: [3][C][1][3][3], bias layout: [3][C]
// ---------------------------------------------------------------------------
__global__ void cpe_kernel(const float* __restrict__ x,
                           const float* __restrict__ w,
                           const float* __restrict__ bias,
                           float* __restrict__ out) {
  const int c  = threadIdx.x;
  const int t  = blockIdx.x;
  const int wq = t % W_;
  const int hq = (t / W_) % H_;
  const int dq = (t / (W_ * H_)) % D_;
  const int bq = t / (W_ * H_ * D_);

  const float* w0 = w + (size_t)0 * C_ * 9 + c * 9;   // hw
  const float* w1 = w + (size_t)1 * C_ * 9 + c * 9;   // dh
  const float* w2 = w + (size_t)2 * C_ * 9 + c * 9;   // dw

  float acc = x[(size_t)t * C_ + c] + bias[c] + bias[C_ + c] + bias[2 * C_ + c];

  // conv over (h, w)
  #pragma unroll
  for (int ky = 0; ky < 3; ++ky) {
    int hh = hq + ky - 1;
    if ((unsigned)hh >= (unsigned)H_) continue;
    #pragma unroll
    for (int kx = 0; kx < 3; ++kx) {
      int wi = wq + kx - 1;
      if ((unsigned)wi >= (unsigned)W_) continue;
      size_t tt = ((size_t)(bq * D_ + dq) * H_ + hh) * W_ + wi;
      acc += x[tt * C_ + c] * w0[ky * 3 + kx];
    }
  }
  // conv over (d, h)
  #pragma unroll
  for (int ky = 0; ky < 3; ++ky) {
    int dd = dq + ky - 1;
    if ((unsigned)dd >= (unsigned)D_) continue;
    #pragma unroll
    for (int kx = 0; kx < 3; ++kx) {
      int hh = hq + kx - 1;
      if ((unsigned)hh >= (unsigned)H_) continue;
      size_t tt = ((size_t)(bq * D_ + dd) * H_ + hh) * W_ + wq;
      acc += x[tt * C_ + c] * w1[ky * 3 + kx];
    }
  }
  // conv over (d, w)
  #pragma unroll
  for (int ky = 0; ky < 3; ++ky) {
    int dd = dq + ky - 1;
    if ((unsigned)dd >= (unsigned)D_) continue;
    #pragma unroll
    for (int kx = 0; kx < 3; ++kx) {
      int wi = wq + kx - 1;
      if ((unsigned)wi >= (unsigned)W_) continue;
      size_t tt = ((size_t)(bq * D_ + dd) * H_ + hq) * W_ + wi;
      acc += x[tt * C_ + c] * w2[ky * 3 + kx];
    }
  }
  out[(size_t)t * C_ + c] = acc;
}

// ---------------------------------------------------------------------------
// LayerNorm over C=256, one wave32 per token, output bf16.
// ---------------------------------------------------------------------------
__global__ void ln_kernel(const float* __restrict__ x,
                          const float* __restrict__ g,
                          const float* __restrict__ b,
                          unsigned short* __restrict__ out) {
  const int lane = threadIdx.x & 31;
  const int t    = blockIdx.x * 8 + (threadIdx.x >> 5);   // T divisible by 8
  const float* row = x + (size_t)t * C_;
  float v[8];
  #pragma unroll
  for (int j = 0; j < 8; ++j) v[j] = row[lane * 8 + j];
  float s = 0.f, s2 = 0.f;
  #pragma unroll
  for (int j = 0; j < 8; ++j) { s += v[j]; s2 += v[j] * v[j]; }
  #pragma unroll
  for (int off = 16; off > 0; off >>= 1) {
    s  += __shfl_xor(s,  off, 32);
    s2 += __shfl_xor(s2, off, 32);
  }
  float mean = s * (1.f / C_);
  float var  = s2 * (1.f / C_) - mean * mean;
  float rstd = rsqrtf(var + 1e-5f);
  unsigned short* orow = out + (size_t)t * C_;
  #pragma unroll
  for (int j = 0; j < 8; ++j) {
    int cidx = lane * 8 + j;
    orow[cidx] = f2b((v[j] - mean) * rstd * g[cidx] + b[cidx]);
  }
}

// ---------------------------------------------------------------------------
// WMMA GEMM:  out[M,N] = A[M,K](bf16) @ W[N,K](bf16)^T + bias[N]   (+ epilogue)
//   mode 0: f32 out
//   mode 1: bf16 out
//   mode 2: bf16 out, exact GELU
//   mode 3: f32 out, += addsrc (residual / accumulation)
// One wave computes a 32(M) x 64(N) tile; K stepped by 32; 8 WMMA per K-step
// (two A fragments share four B fragments -> ~21 flop/byte per lane).
// Requires M%32==0, N%64==0, K%32==0 (true for every call here).
// ---------------------------------------------------------------------------
__global__ __launch_bounds__(256)
void gemm_bf16_wmma(const unsigned short* __restrict__ A,
                    const unsigned short* __restrict__ Wt,
                    const float* __restrict__ bias,
                    const float* __restrict__ addsrc,
                    void* __restrict__ outp,
                    int M, int N, int K, int mode) {
  const int lane   = threadIdx.x & 31;
  const int gw     = blockIdx.x * 8 + (threadIdx.x >> 5);
  const int nT     = N >> 6;
  const int totalW = (M >> 5) * nT;
  if (gw >= totalW) return;                 // wave-uniform exit, EXEC stays full
  const int mt   = gw / nT;
  const int nt   = gw - mt * nT;
  const int half = lane >> 4;               // lane group 0/1
  const int li   = lane & 15;

  // A fragments: rows mt*32+li and mt*32+16+li ; K elems {half*8..+7} and {16+half*8..+7}
  const unsigned short* Abase0 = A + (size_t)(mt * 32 + li) * K + half * 8;
  const unsigned short* Abase1 = Abase0 + (size_t)16 * K;
  // B fragment: col = nt*64 + s*16 + li ; K elems {half*8..+7} and {16+half*8..+7}
  const int bcol0 = nt * 64 + li;
  const unsigned short* Bbase = Wt + half * 8;

  v8f acc[2][4] = {};
  union FU { uint4 u[2]; v16bf v; };

  for (int kb = 0; kb < K; kb += 32) {
    FU fa0, fa1;
    const uint4* pa0 = (const uint4*)(Abase0 + kb);
    const uint4* pa1 = (const uint4*)(Abase1 + kb);
    fa0.u[0] = pa0[0];          // K: kb+half*8    .. +7
    fa0.u[1] = pa0[2];          // K: kb+16+half*8 .. +7
    fa1.u[0] = pa1[0];
    fa1.u[1] = pa1[2];
    if (kb + 32 < K) {          // prefetch next A K-slab (global_prefetch_b8)
      __builtin_prefetch(Abase0 + kb + 32, 0, 1);
      __builtin_prefetch(Abase1 + kb + 32, 0, 1);
    }
    #pragma unroll
    for (int s = 0; s < 4; ++s) {
      FU fb;
      const uint4* pb = (const uint4*)(Bbase + (size_t)(bcol0 + s * 16) * K + kb);
      fb.u[0] = pb[0];
      fb.u[1] = pb[2];
      acc[0][s] = __builtin_amdgcn_wmma_f32_16x16x32_bf16(
          false, fa0.v, false, fb.v, (short)0, acc[0][s], false, false);
      acc[1][s] = __builtin_amdgcn_wmma_f32_16x16x32_bf16(
          false, fa1.v, false, fb.v, (short)0, acc[1][s], false, false);
    }
  }

  #pragma unroll
  for (int a = 0; a < 2; ++a) {
    #pragma unroll
    for (int s = 0; s < 4; ++s) {
      #pragma unroll
      for (int r = 0; r < 8; ++r) {
        int row  = mt * 32 + a * 16 + r + 8 * half;  // C/D layout: VGPR r -> M=r (+8 hi lanes)
        int coln = nt * 64 + s * 16 + li;
        float v  = acc[a][s][r] + bias[coln];
        size_t idx = (size_t)row * N + coln;
        if (mode == 0) {
          ((float*)outp)[idx] = v;
        } else if (mode == 1) {
          ((unsigned short*)outp)[idx] = f2b(v);
        } else if (mode == 2) {
          v = 0.5f * v * (1.f + erff(v * 0.70710678118654752f));
          ((unsigned short*)outp)[idx] = f2b(v);
        } else {
          ((float*)outp)[idx] = addsrc[idx] + v;
        }
      }
    }
  }
}

// ---------------------------------------------------------------------------
// Window token index maps (no padding: 8%2==0, 56%7==0).
// branch 0: xy windows 7x7  (per b,d,hb,wb)       nw = 4*8*64   = 2048, n = 49
// branch 1: ty windows 2x7  (per b,db,hb,w)       nw = 4*4*8*56 = 7168, n = 14
// branch 2: tx windows 2x7  (per b,db,h,wb)       nw = 4*4*56*8 = 7168, n = 14
// ---------------------------------------------------------------------------
__device__ __forceinline__ int win_token(int branch, int wi, int e) {
  int b, d, h, w;
  if (branch == 0) {
    b = wi / (D_ * 64);
    int rem = wi % (D_ * 64);
    d = rem / 64;
    int blk = rem % 64;
    h = (blk >> 3) * 7 + e / 7;
    w = (blk & 7) * 7 + e % 7;
  } else if (branch == 1) {
    b = wi / (4 * 8 * W_);
    int rem = wi % (4 * 8 * W_);
    int db  = rem / (8 * W_);
    int r2  = rem % (8 * W_);
    int hb  = r2 / W_;
    w = r2 % W_;
    d = db * 2 + e / 7;
    h = hb * 7 + e % 7;
  } else {
    b = wi / (4 * H_ * 8);
    int rem = wi % (4 * H_ * 8);
    int db  = rem / (H_ * 8);
    int r2  = rem % (H_ * 8);
    h = r2 / 8;
    d = db * 2 + e / 7;
    w = (r2 & 7) * 7 + e % 7;
  }
  return ((b * D_ + d) * H_ + h) * W_ + w;
}

// ---------------------------------------------------------------------------
// Window attention. One block per window; window's qkv rows (n x 768 bf16)
// staged into LDS (async global->LDS where available, ASYNCcnt-tracked);
// one thread per (head, query-row) with online softmax.
// qkv row layout: [q(8x32) | k(8x32) | v(8x32)] = 768 bf16.
// ---------------------------------------------------------------------------
__global__ void attn_kernel(const unsigned short* __restrict__ qkv,
                            unsigned short* __restrict__ outb,
                            int branch, int n) {
  extern __shared__ unsigned short sq[];          // [n][768]
  const int wi = blockIdx.x;

  const int chunksPerRow = 768 / 8;               // 16-byte chunks per row
  const int nchunks = n * chunksPerRow;
  for (int i = threadIdx.x; i < nchunks; i += blockDim.x) {
    int r   = i / chunksPerRow;
    int col = (i - r * chunksPerRow) * 8;
    int t   = win_token(branch, wi, r);
    const unsigned short* gsrc = qkv + (size_t)t * 768 + col;
    unsigned short*       ldst = sq + (size_t)i * 8;   // == r*768 + col
#if HAVE_ASYNC_LDS
    __builtin_amdgcn_global_load_async_to_lds_b128(
        (gv4i_p)gsrc, (lv4i_p)ldst, 0, 0);
#else
    *(uint4*)ldst = *(const uint4*)gsrc;
#endif
  }
  wait_async_lds();
  __syncthreads();

  const int total = HEADS_ * n;
  for (int item = threadIdx.x; item < total; item += blockDim.x) {
    const int h = item / n;
    const int r = item - h * n;
    const unsigned short* qp = sq + r * 768 + h * HD_;
    float qv[HD_];
    #pragma unroll
    for (int d = 0; d < HD_; ++d)
      qv[d] = b2f(qp[d]) * 0.17677669529663689f;   // 1/sqrt(32)

    float m = -1e30f, l = 0.f;
    float o[HD_];
    #pragma unroll
    for (int d = 0; d < HD_; ++d) o[d] = 0.f;

    for (int i = 0; i < n; ++i) {
      const unsigned short* kp = sq + i * 768 + 256 + h * HD_;
      float s = 0.f;
      #pragma unroll
      for (int d = 0; d < HD_; ++d) s += qv[d] * b2f(kp[d]);
      float mn   = fmaxf(m, s);
      float corr = __expf(m - mn);
      float p    = __expf(s - mn);
      l = l * corr + p;
      const unsigned short* vp = sq + i * 768 + 512 + h * HD_;
      #pragma unroll
      for (int d = 0; d < HD_; ++d) o[d] = o[d] * corr + p * b2f(vp[d]);
      m = mn;
    }
    float inv = 1.f / l;
    int ot = win_token(branch, wi, r);
    unsigned short* op = outb + (size_t)ot * C_ + h * HD_;
    #pragma unroll
    for (int d = 0; d < HD_; ++d) op[d] = f2b(o[d] * inv);
  }
}

// ---------------------------------------------------------------------------
// Host orchestration
// ---------------------------------------------------------------------------
extern "C" void kernel_launch(void* const* d_in, const int* in_sizes, int n_in,
                              void* d_out, int out_size, void* d_ws, size_t ws_size,
                              hipStream_t stream) {
  const float* x      = (const float*)d_in[0];
  const float* cpe1_w = (const float*)d_in[1];
  const float* cpe1_b = (const float*)d_in[2];
  const float* cpe2_w = (const float*)d_in[3];
  const float* cpe2_b = (const float*)d_in[4];
  const float* ln1_g  = (const float*)d_in[5];
  const float* ln1_b  = (const float*)d_in[6];
  const float* qkv_w  = (const float*)d_in[7];
  const float* qkv_b  = (const float*)d_in[8];
  const float* proj_w = (const float*)d_in[9];
  const float* proj_b = (const float*)d_in[10];
  const float* ln2_g  = (const float*)d_in[11];
  const float* ln2_b  = (const float*)d_in[12];
  const float* fc1_w  = (const float*)d_in[13];
  const float* fc1_b  = (const float*)d_in[14];
  const float* fc2_w  = (const float*)d_in[15];
  const float* fc2_b  = (const float*)d_in[16];
  float* out = (float*)d_out;
  (void)in_sizes; (void)n_in; (void)out_size; (void)ws_size;

  char* ws = (char*)d_ws;
  size_t off = 0;
  auto carve = [&](size_t bytes) -> char* {
    char* p = ws + off;
    off += (bytes + 255) & ~(size_t)255;
    return p;
  };
  unsigned short* wqkv  = (unsigned short*)carve((size_t)3 * 768 * 256 * 2);
  unsigned short* wproj = (unsigned short*)carve((size_t)3 * 256 * 256 * 2);
  unsigned short* wfc1  = (unsigned short*)carve((size_t)HID_ * 256 * 2);
  unsigned short* wfc2  = (unsigned short*)carve((size_t)256 * HID_ * 2);
  float*          x1f   = (float*)carve((size_t)T_ * C_ * 4);
  float*          accf  = (float*)carve((size_t)T_ * C_ * 4);
  unsigned short* curb  = (unsigned short*)carve((size_t)T_ * C_ * 2);
  unsigned short* bigb  = (unsigned short*)carve((size_t)T_ * HID_ * 2); // qkv / mlp hidden
  unsigned short* attb  = (unsigned short*)carve((size_t)T_ * C_ * 2);

  // Weight conversion to bf16 (cheap, re-done every call: deterministic)
  { int n = 3 * 768 * 256; cvt_f32_bf16<<<(n + 255) / 256, 256, 0, stream>>>(qkv_w,  wqkv,  n); }
  { int n = 3 * 256 * 256; cvt_f32_bf16<<<(n + 255) / 256, 256, 0, stream>>>(proj_w, wproj, n); }
  { int n = HID_ * 256;    cvt_f32_bf16<<<(n + 255) / 256, 256, 0, stream>>>(fc1_w,  wfc1,  n); }
  { int n = 256 * HID_;    cvt_f32_bf16<<<(n + 255) / 256, 256, 0, stream>>>(fc2_w,  wfc2,  n); }

  // x = cpe1(x)
  cpe_kernel<<<T_, 256, 0, stream>>>(x, cpe1_w, cpe1_b, x1f);
  // cur = ln1(x)
  ln_kernel<<<T_ / 8, 256, 0, stream>>>(x1f, ln1_g, ln1_b, curb);

  const int nw[3] = {B_ * D_ * 64, B_ * 4 * 8 * W_, B_ * 4 * H_ * 8};
  const int nn[3] = {49, 14, 14};
  for (int br = 0; br < 3; ++br) {
    // qkv = cur @ Wqkv^T + b   -> bf16 [T, 768]
    { int tw = (T_ / 32) * (768 / 64);
      gemm_bf16_wmma<<<(tw + 7) / 8, 256, 0, stream>>>(
          curb, wqkv + (size_t)br * 768 * 256, qkv_b + br * 768,
          nullptr, bigb, T_, 768, 256, 1); }
    // window attention -> bf16 [T, 256]
    attn_kernel<<<nw[br], 256, (size_t)nn[br] * 768 * 2, stream>>>(
        bigb, attb, br, nn[br]);
    // acc = (br==0 ? x1 : acc) + att @ Wproj^T + b
    { int tw = (T_ / 32) * (256 / 64);
      gemm_bf16_wmma<<<(tw + 7) / 8, 256, 0, stream>>>(
          attb, wproj + (size_t)br * 256 * 256, proj_b + br * 256,
          (br == 0) ? x1f : accf, accf, T_, 256, 256, 3); }
  }

  // x = cpe2(x)
  cpe_kernel<<<T_, 256, 0, stream>>>(accf, cpe2_w, cpe2_b, x1f);
  // h = ln2(x)
  ln_kernel<<<T_ / 8, 256, 0, stream>>>(x1f, ln2_g, ln2_b, curb);
  // hid = gelu(h @ fc1^T + b)  -> bf16 [T, 1024]
  { int tw = (T_ / 32) * (HID_ / 64);
    gemm_bf16_wmma<<<(tw + 7) / 8, 256, 0, stream>>>(
        curb, wfc1, fc1_b, nullptr, bigb, T_, HID_, 256, 2); }
  // out = x + hid @ fc2^T + b
  { int tw = (T_ / 32) * (256 / 64);
    gemm_bf16_wmma<<<(tw + 7) / 8, 256, 0, stream>>>(
        bigb, wfc2, fc2_b, x1f, out, T_, 256, HID_, 3); }
}
